// OutputDecoder_18210661335282
// MI455X (gfx1250) — matrix operand order repaired and checked by reading, not compiled, not run on hardware
//
#include <hip/hip_runtime.h>
#include <math.h>

typedef __attribute__((ext_vector_type(16))) __bf16 v16bf;
typedef __attribute__((ext_vector_type(8)))  __bf16 v8bf;
typedef __attribute__((ext_vector_type(8)))  float  v8f;
typedef __attribute__((ext_vector_type(4)))  int    v4i;

typedef __attribute__((address_space(1))) v4i gv4i;   // global int4
typedef __attribute__((address_space(3))) v4i sv4i;   // LDS int4

#define DEV static __device__ __forceinline__

constexpr int V_ = 32000, E_ = 256, H_ = 512, B_ = 64, T_ = 2048;
constexpr int G4H = 2048; // 4*H
constexpr int EP  = 520;  // padded LDS row stride (floats): 2080B, 32B-aligned, bank-shifted

DEV v8f wmma_bf16(v16bf a, v16bf b, v8f c) {
#if defined(__gfx1250__)
  return __builtin_amdgcn_wmma_f32_16x16x32_bf16(false, a, false, b, (short)0, c,
                                                 false, false);
#else
  (void)a; (void)b; return c;
#endif
}

DEV float sigm(float x) { return 1.0f / (1.0f + expf(-x)); }

// ---------------- prep: Wk (attention key weights) f32 -> bf16 ----------------
__global__ void k_prep_wk(const float* __restrict__ Wattn, __bf16* __restrict__ wk) {
  int idx = blockIdx.x * 256 + threadIdx.x;     // 512*512
  int k = idx >> 9, h = idx & 511;
  wk[idx] = (__bf16)Wattn[(size_t)k * 1024 + 512 + h];
}

// ---------------- prep: hb[b,h] = h_top @ Wq^T + b_attn ----------------
__global__ void k_hb(const float* __restrict__ htop, const float* __restrict__ Wattn,
                     const float* __restrict__ battn, float* __restrict__ hb) {
  int idx = blockIdx.x * 256 + threadIdx.x;     // 64*512
  int b = idx >> 9, i = idx & 511;
  float acc = battn[i];
  const float* hr = htop + b * H_;
  const float* wr = Wattn + (size_t)i * 1024;   // first half of row i
  for (int k = 0; k < H_; ++k) acc += hr[k] * wr[k];
  hb[idx] = acc;
}

// ---------------- fused attention-energy GEMM (bf16 WMMA) ----------------
// energy[b,t] = sum_h tanh(hb[b,h] + (enc[b,t,:] . Wk[h,:])) * v[h]
// WG = 4 waves, covers 32 t-rows of one batch b; wave w owns h-tiles w*8..w*8+7.
// kt outer (A built once per kt), ht/mt inner; launch_bounds(128,1) gives the
// full VGPR budget so all 16 accumulator tiles (128 VGPRs) stay in registers.
__global__ void __launch_bounds__(128, 1)
k_energy(const float* __restrict__ enc, const __bf16* __restrict__ wk,
         const float* __restrict__ hb, const float* __restrict__ vattn,
         float* __restrict__ energy) {
  __shared__ float sEnc[32 * EP];   // 32 rows x 512 f32, padded; ~65 KB
  __shared__ float sE[32];
  const int b   = blockIdx.y;
  const int t0  = blockIdx.x * 32;
  const int tid = threadIdx.x;

  const float* src = enc + ((size_t)b * T_ + t0) * H_;   // 16384 contiguous f32

#if __has_builtin(__builtin_amdgcn_global_load_async_to_lds_b128)
  // CDNA5 async memory->LDS copy (tracked by ASYNCcnt), 16B per lane per op.
  for (int i = 0; i < 32; ++i) {
    int c = i * 128 + tid;            // 4096 chunks of 16B
    int row = c >> 7, col4 = (c & 127) * 4;
    __builtin_amdgcn_global_load_async_to_lds_b128(
        (gv4i*)(src + (size_t)row * H_ + col4),
        (sv4i*)(&sEnc[row * EP + col4]), 0, 0);
  }
#if __has_builtin(__builtin_amdgcn_s_wait_asynccnt)
  __builtin_amdgcn_s_wait_asynccnt(0);
#else
  asm volatile("s_wait_asynccnt 0" ::: "memory");
#endif
#else
  for (int i = 0; i < 128; ++i) {
    int idx = i * 128 + tid;          // 16384 floats
    int row = idx >> 9, col = idx & 511;
    sEnc[row * EP + col] = src[idx];
  }
#endif
  if (tid < 32) sE[tid] = 0.f;
  __syncthreads();

  const int w = tid >> 5, lane = tid & 31, hi = lane >> 4, ln = lane & 15;

  v8f c[2][8];
#pragma unroll
  for (int mt = 0; mt < 2; ++mt)
#pragma unroll
    for (int ht = 0; ht < 8; ++ht)
#pragma unroll
      for (int j = 0; j < 8; ++j) c[mt][ht][j] = 0.f;

  for (int kt = 0; kt < 16; ++kt) {   // rolled: keeps I$ small, all operands per-kt
    const int kb = kt * 32;
    v16bf A[2];
#pragma unroll
    for (int mt = 0; mt < 2; ++mt) {
      const int base = (mt * 16 + ln) * EP + kb;
      v8f flo = *(const v8f*)&sEnc[base + 16 * hi];
      v8f fhi = *(const v8f*)&sEnc[base + 16 * hi + 8];
#pragma unroll
      for (int j = 0; j < 8; ++j) {
        A[mt][j]     = (__bf16)flo[j];
        A[mt][j + 8] = (__bf16)fhi[j];
      }
    }
#pragma unroll
    for (int ht = 0; ht < 8; ++ht) {
      const int hbase = (w * 8 + ht) * 16;
      v16bf Bf = *(const v16bf*)&wk[(size_t)(hbase + ln) * H_ + kb + 16 * hi];
      c[0][ht] = wmma_bf16(A[0], Bf, c[0][ht]);
      c[1][ht] = wmma_bf16(A[1], Bf, c[1][ht]);
    }
  }

  float acc[2][8];
#pragma unroll
  for (int mt = 0; mt < 2; ++mt)
#pragma unroll
    for (int r = 0; r < 8; ++r) acc[mt][r] = 0.f;

#pragma unroll
  for (int ht = 0; ht < 8; ++ht) {
    const int hcol = (w * 8 + ht) * 16 + ln;
    const float hbv = hb[b * H_ + hcol];
    const float vv  = vattn[hcol];
#pragma unroll
    for (int mt = 0; mt < 2; ++mt)
#pragma unroll
      for (int r = 0; r < 8; ++r)
        acc[mt][r] += tanhf(hbv + c[mt][ht][r]) * vv;
  }
#pragma unroll
  for (int mt = 0; mt < 2; ++mt)
#pragma unroll
    for (int r = 0; r < 8; ++r) {
      float v = acc[mt][r];
      v += __shfl_xor(v, 1, 32);
      v += __shfl_xor(v, 2, 32);
      v += __shfl_xor(v, 4, 32);
      v += __shfl_xor(v, 8, 32);
      if (ln == 0) atomicAdd(&sE[mt * 16 + r + 8 * hi], v);
    }
  __syncthreads();
  if (tid < 32) energy[(size_t)b * T_ + t0 + tid] = sE[tid];
}

// ---------------- masked softmax over T ----------------
__global__ void k_softmax(const float* __restrict__ energy, const int* __restrict__ mask,
                          float* __restrict__ attw) {
  __shared__ float red[256];
  const int b = blockIdx.x, tid = threadIdx.x;
  float e[8];
  float mx = -__builtin_huge_valf();
#pragma unroll
  for (int i = 0; i < 8; ++i) {
    int t = i * 256 + tid;
    float v = energy[b * T_ + t];
    if (mask[b * T_ + t] == 0) v = -__builtin_huge_valf();
    e[i] = v; mx = fmaxf(mx, v);
  }
  red[tid] = mx; __syncthreads();
  for (int s = 128; s > 0; s >>= 1) { if (tid < s) red[tid] = fmaxf(red[tid], red[tid + s]); __syncthreads(); }
  mx = red[0]; __syncthreads();
  float sum = 0.f;
#pragma unroll
  for (int i = 0; i < 8; ++i) { e[i] = expf(e[i] - mx); sum += e[i]; }
  red[tid] = sum; __syncthreads();
  for (int s = 128; s > 0; s >>= 1) { if (tid < s) red[tid] += red[tid + s]; __syncthreads(); }
  float inv = 1.0f / red[0];
#pragma unroll
  for (int i = 0; i < 8; ++i) attw[b * T_ + i * 256 + tid] = e[i] * inv;
}

// ---------------- context = attn_weights . enc ----------------
__global__ void k_context(const float* __restrict__ enc, const float* __restrict__ attw,
                          float* __restrict__ ctx) {
  const int b = blockIdx.x, tid = threadIdx.x;
  float a0 = 0.f, a1 = 0.f;
  const float* eb = enc + (size_t)b * T_ * H_;
  const float* wb = attw + b * T_;
  for (int t = 0; t < T_; ++t) {
    float wv = wb[t];
    a0 += wv * eb[(size_t)t * H_ + tid];
    a1 += wv * eb[(size_t)t * H_ + tid + 256];
  }
  ctx[b * H_ + tid] = a0;
  ctx[b * H_ + tid + 256] = a1;
}

// ---------------- x0 = concat(embedded, context) ----------------
__global__ void k_build_x0(const int* __restrict__ token, const float* __restrict__ emb,
                           const float* __restrict__ ctx, float* __restrict__ x0) {
  const int b = blockIdx.x, tid = threadIdx.x;
  const int tok = token[b];
#pragma unroll
  for (int i = 0; i < 3; ++i) {
    int c = i * 256 + tid;
    float v = (c < E_) ? emb[(size_t)tok * E_ + c] : ctx[b * H_ + (c - E_)];
    x0[b * 768 + c] = v;
  }
}

// ---------------- LSTM gates GEMM: [64,Kx]@Wi^T + [64,512]@Wh^T (bf16 WMMA) ----------------
__global__ void k_gates(const float* __restrict__ x, int Kx,
                        const float* __restrict__ hprev,
                        const float* __restrict__ Wi, const float* __restrict__ Wh,
                        const float* __restrict__ bi, const float* __restrict__ bh,
                        float* __restrict__ gates) {
  __shared__ __bf16 sS[64 * 128];
  const int tid = threadIdx.x;
  const int w = tid >> 5, lane = tid & 31, hi = lane >> 4, ln = lane & 15;
  const int n0w = blockIdx.x * 64 + w * 16;
  v8f c[4];
#pragma unroll
  for (int m = 0; m < 4; ++m)
#pragma unroll
    for (int j = 0; j < 8; ++j) c[m][j] = 0.f;

  const int nkx = Kx / 128;
  const int nkc = nkx + 4;   // Kh = 512 -> 4 chunks
  for (int kc = 0; kc < nkc; ++kc) {
    const bool isx = kc < nkx;
    const float* src = isx ? x : hprev;
    const float* W   = isx ? Wi : Wh;
    const int K   = isx ? Kx : 512;
    const int kb0 = (isx ? kc : (kc - nkx)) * 128;
    for (int i = 0; i < 64; ++i) {           // stage 64x128 activations as bf16
      int idx = i * 128 + tid;
      int row = idx >> 7, col = idx & 127;
      sS[idx] = (__bf16)src[(size_t)row * K + kb0 + col];
    }
    __syncthreads();
#pragma unroll
    for (int kt = 0; kt < 4; ++kt) {
      const int kl = kt * 32;
      const float* wp = W + (size_t)(n0w + ln) * K + kb0 + kl + 16 * hi;
      v8f f0 = *(const v8f*)wp;
      v8f f1 = *(const v8f*)(wp + 8);
      v16bf Bf;
#pragma unroll
      for (int j = 0; j < 8; ++j) { Bf[j] = (__bf16)f0[j]; Bf[j + 8] = (__bf16)f1[j]; }
#pragma unroll
      for (int mt = 0; mt < 4; ++mt) {
        int row = mt * 16 + ln;
        v8bf lo = *(const v8bf*)&sS[row * 128 + kl + 16 * hi];
        v8bf h8 = *(const v8bf*)&sS[row * 128 + kl + 16 * hi + 8];
        v16bf A;
#pragma unroll
        for (int j = 0; j < 8; ++j) { A[j] = lo[j]; A[j + 8] = h8[j]; }
        c[mt] = wmma_bf16(A, Bf, c[mt]);
      }
    }
    __syncthreads();
  }
  const int n = n0w + ln;
  const float bsum = bi[n] + bh[n];
#pragma unroll
  for (int mt = 0; mt < 4; ++mt)
#pragma unroll
    for (int r = 0; r < 8; ++r) {
      int bb = mt * 16 + r + 8 * hi;
      gates[(size_t)bb * G4H + n] = c[mt][r] + bsum;
    }
}

// ---------------- LSTM elementwise activations ----------------
__global__ void k_lstm_act(const float* __restrict__ gates, const float* __restrict__ cprev,
                           float* __restrict__ hout, float* __restrict__ cout) {
  int idx = blockIdx.x * 256 + threadIdx.x;   // B*H
  int b = idx >> 9, u = idx & 511;
  float gi = gates[(size_t)b * G4H + u];
  float gf = gates[(size_t)b * G4H + u + 512];
  float gg = gates[(size_t)b * G4H + u + 1024];
  float go = gates[(size_t)b * G4H + u + 1536];
  float cn = sigm(gf) * cprev[idx] + sigm(gi) * tanhf(gg);
  hout[idx] = sigm(go) * tanhf(cn);
  cout[idx] = cn;
}

// ---------------- logits = h1 @ W_out^T + b_out (bf16 WMMA, W_out read once) ----------------
__global__ void k_logits(const float* __restrict__ h1, const float* __restrict__ Wout,
                         const float* __restrict__ bout, float* __restrict__ logits) {
  extern __shared__ __bf16 sH[];   // 64*512 bf16 = 64 KB
  const int tid = threadIdx.x;
  for (int i = 0; i < 256; ++i) {
    int idx = i * 128 + tid;
    sH[idx] = (__bf16)h1[idx];
  }
  __syncthreads();
  const int w = tid >> 5, lane = tid & 31, hi = lane >> 4, ln = lane & 15;
  const int n0w = blockIdx.x * 64 + w * 16;
  const int n = n0w + ln;
  v8f c[4];
#pragma unroll
  for (int m = 0; m < 4; ++m)
#pragma unroll
    for (int j = 0; j < 8; ++j) c[m][j] = 0.f;
#pragma unroll
  for (int kt = 0; kt < 16; ++kt) {
    const int kb = kt * 32;
    const float* wp = Wout + (size_t)n * H_ + kb + 16 * hi;
    __builtin_prefetch(wp + 32, 0, 1);   // next kt chunk of this row -> global_prefetch_b8
    v8f f0 = *(const v8f*)wp;
    v8f f1 = *(const v8f*)(wp + 8);
    v16bf Bf;
#pragma unroll
    for (int j = 0; j < 8; ++j) { Bf[j] = (__bf16)f0[j]; Bf[j + 8] = (__bf16)f1[j]; }
#pragma unroll
    for (int mt = 0; mt < 4; ++mt) {
      int row = mt * 16 + ln;
      v8bf lo = *(const v8bf*)&sH[row * 512 + kb + 16 * hi];
      v8bf h8 = *(const v8bf*)&sH[row * 512 + kb + 16 * hi + 8];
      v16bf A;
#pragma unroll
      for (int j = 0; j < 8; ++j) { A[j] = lo[j]; A[j + 8] = h8[j]; }
      c[mt] = wmma_bf16(A, Bf, c[mt]);
    }
  }
  const float bo = bout[n];
#pragma unroll
  for (int mt = 0; mt < 4; ++mt)
#pragma unroll
    for (int r = 0; r < 8; ++r) {
      int bb = mt * 16 + r + 8 * hi;
      logits[(size_t)bb * V_ + n] = c[mt][r] + bo;
    }
}

extern "C" void kernel_launch(void* const* d_in, const int* in_sizes, int n_in,
                              void* d_out, int out_size, void* d_ws, size_t ws_size,
                              hipStream_t stream) {
  const int*   token = (const int*)d_in[0];
  const float* h0    = (const float*)d_in[1];
  const float* c0    = (const float*)d_in[2];
  const float* enc   = (const float*)d_in[3];
  const int*   mask  = (const int*)d_in[4];
  const float* emb   = (const float*)d_in[5];
  const float* Wattn = (const float*)d_in[6];
  const float* battn = (const float*)d_in[7];
  const float* vattn = (const float*)d_in[8];
  const float* Wih0  = (const float*)d_in[9];
  const float* Whh0  = (const float*)d_in[10];
  const float* bih0  = (const float*)d_in[11];
  const float* bhh0  = (const float*)d_in[12];
  const float* Wih1  = (const float*)d_in[13];
  const float* Whh1  = (const float*)d_in[14];
  const float* bih1  = (const float*)d_in[15];
  const float* bhh1  = (const float*)d_in[16];
  const float* Wout  = (const float*)d_in[17];
  const float* bout  = (const float*)d_in[18];

  float* out    = (float*)d_out;
  float* logits = out;                              // [B,V]
  float* hnew   = out + (size_t)B_ * V_;            // [2,B,H]
  float* cnew   = hnew + 2 * B_ * H_;               // [2,B,H]
  float* attw   = cnew + 2 * B_ * H_;               // [B,T]

  char* ws = (char*)d_ws;
  __bf16* wk    = (__bf16*)ws;                 // 512 KB
  float*  hb    = (float*)(ws + 524288);       // 128 KB
  float*  energy= (float*)(ws + 655360);       // 512 KB
  float*  ctx   = (float*)(ws + 1179648);      // 128 KB
  float*  x0    = (float*)(ws + 1310720);      // 192 KB
  float*  gates = (float*)(ws + 1507328);      // 512 KB

  const float* htop = h0 + (size_t)B_ * H_;    // top-layer hidden

  k_prep_wk<<<dim3(1024), dim3(256), 0, stream>>>(Wattn, wk);
  k_hb<<<dim3(128), dim3(256), 0, stream>>>(htop, Wattn, battn, hb);
  k_energy<<<dim3(T_ / 32, B_), dim3(128), 0, stream>>>(enc, wk, hb, vattn, energy);
  k_softmax<<<dim3(B_), dim3(256), 0, stream>>>(energy, mask, attw);
  k_context<<<dim3(B_), dim3(256), 0, stream>>>(enc, attw, ctx);
  k_build_x0<<<dim3(B_), dim3(256), 0, stream>>>(token, emb, ctx, x0);
  // LSTM layer 0
  k_gates<<<dim3(32), dim3(128), 0, stream>>>(x0, 768, h0, Wih0, Whh0, bih0, bhh0, gates);
  k_lstm_act<<<dim3(128), dim3(256), 0, stream>>>(gates, c0, hnew, cnew);
  // LSTM layer 1
  k_gates<<<dim3(32), dim3(128), 0, stream>>>(hnew, 512, htop, Wih1, Whh1, bih1, bhh1, gates);
  k_lstm_act<<<dim3(128), dim3(256), 0, stream>>>(gates, c0 + B_ * H_,
                                                  hnew + B_ * H_, cnew + B_ * H_);
  // output projection
  k_logits<<<dim3(V_ / 64), dim3(128), 64 * 512 * sizeof(__bf16), stream>>>(
      hnew + B_ * H_, Wout, bout, logits);
}
